// AttentionLayers_18915035972050
// MI455X (gfx1250) — matrix operand order repaired and checked by reading, not compiled, not run on hardware
//
#include <hip/hip_runtime.h>
#include <hip/hip_bf16.h>

// ---------------------------------------------------------------------------
// AttentionLayers for MI455X (gfx1250, wave32, WMMA).
// All GEMM classes run on v_wmma_f32_16x16x32_f16 with LDS-staged 128x64x32
// tiles (4 waves, each wave 64x32 = 8 WMMA per K-step). FULL-tile kernels
// (projection GEMMs) use async global->LDS staging (ASYNCcnt path).
// ---------------------------------------------------------------------------

typedef _Float16 half_t;
typedef __attribute__((ext_vector_type(16))) _Float16 v16h;
typedef __attribute__((ext_vector_type(8)))  _Float16 v8h;
typedef __attribute__((ext_vector_type(8)))  float    v8f;
typedef __attribute__((ext_vector_type(4)))  int      v4i;

#define B_    2
#define N_    1024
#define NCTX_ 2048
#define DIM_  1024
#define H_    16
#define DH_   64
#define NMEM_ 16
#define J_    (NMEM_ + NCTX_)   // 2064

// gfx1250 async global->LDS staging (ASYNCcnt path), if the toolchain has it.
#if defined(__HIP_DEVICE_COMPILE__)
#if __has_builtin(__builtin_amdgcn_global_load_async_to_lds_b128) && \
    __has_builtin(__builtin_amdgcn_s_wait_asynccnt)
#define USE_ASYNC 1
#endif
#endif
#ifndef USE_ASYNC
#define USE_ASYNC 0
#endif

#if USE_ASYNC
// prototype (from hipcc diagnostic): (v4i AS1*, v4i AS3*, imm int, imm int)
typedef __attribute__((address_space(1))) v4i* as1_v4i_p;
typedef __attribute__((address_space(3))) v4i* as3_v4i_p;
__device__ static inline void async_copy_b128(const half_t* g, half_t* l) {
    __builtin_amdgcn_global_load_async_to_lds_b128(
        (as1_v4i_p)(g), (as3_v4i_p)(l), 0, 0);
}
#endif

// ---------------- tiled batched NT GEMM: C[M][N] = A[M][K] * Bt[N][K]^T -----
#define GBM 128
#define GBN 64
#define GBK 32
#define LDS_S 40   // halfs; multiple of 8 (16B-aligned v8h loads), conflict-breaking

template <bool FULL>
__global__ __launch_bounds__(128)
void gemm_nt_kernel(const half_t* __restrict__ A, const half_t* __restrict__ Bt,
                    float* __restrict__ C,
                    int M, int N, int K,
                    long sA, long sB, long sC,
                    int lda, int ldb, int ldc, int beta)
{
    __shared__ __align__(16) half_t As[GBM * LDS_S];
    __shared__ __align__(16) half_t Bs[GBN * LDS_S];

    const int batch = blockIdx.z;
    A  += (long)batch * sA;
    Bt += (long)batch * sB;
    C  += (long)batch * sC;

    const int tileN = blockIdx.x * GBN;
    const int tileM = blockIdx.y * GBM;

    const int tid   = threadIdx.x;       // 0..127 (4 waves)
    const int wave  = tid >> 5;
    const int lane  = tid & 31;
    const int waveM = (wave >> 1) * 64;  // 0 | 64
    const int waveN = (wave & 1) * 32;   // 0 | 32
    const int lrow  = lane & 15;
    const int hsel  = (lane >> 4) & 1;   // lanes 0-15 vs 16-31

    v8f acc[4][2] = {};
    union F16x16 { v16h v; v8h h[2]; };

    for (int k0 = 0; k0 < K; k0 += GBK) {
        // uniform trip counts: no divergent compares -> no exec-mask unswitching
        if (FULL) {
#if USE_ASYNC
            #pragma unroll
            for (int i = 0; i < GBM * (GBK / 8) / 128; ++i) {     // 4 iters
                int c = tid + i * 128;
                int r = c >> 2, kk = (c & 3) * 8;
                async_copy_b128(A + (long)(tileM + r) * lda + k0 + kk,
                                As + r * LDS_S + kk);
            }
            #pragma unroll
            for (int i = 0; i < GBN * (GBK / 8) / 128; ++i) {     // 2 iters
                int c = tid + i * 128;
                int r = c >> 2, kk = (c & 3) * 8;
                async_copy_b128(Bt + (long)(tileN + r) * ldb + k0 + kk,
                                Bs + r * LDS_S + kk);
            }
            __builtin_amdgcn_s_wait_asynccnt(0);
#else
            #pragma unroll
            for (int i = 0; i < GBM * (GBK / 8) / 128; ++i) {
                int c = tid + i * 128;
                int r = c >> 2, kk = (c & 3) * 8;
                *(v8h*)(As + r * LDS_S + kk) =
                    *(const v8h*)(A + (long)(tileM + r) * lda + k0 + kk);
            }
            #pragma unroll
            for (int i = 0; i < GBN * (GBK / 8) / 128; ++i) {
                int c = tid + i * 128;
                int r = c >> 2, kk = (c & 3) * 8;
                *(v8h*)(Bs + r * LDS_S + kk) =
                    *(const v8h*)(Bt + (long)(tileN + r) * ldb + k0 + kk);
            }
#endif
        } else {
            // guarded: 8-half granularity vector-load-or-zero (K,lda,ldb % 8 == 0)
            #pragma unroll
            for (int i = 0; i < GBM * (GBK / 8) / 128; ++i) {
                int c = tid + i * 128;
                int r = c >> 2, kk = (c & 3) * 8;
                int gr = tileM + r, gk = k0 + kk;
                v8h val = {};
                if (gr < M && gk < K)
                    val = *(const v8h*)(A + (long)gr * lda + gk);
                *(v8h*)(As + r * LDS_S + kk) = val;
            }
            #pragma unroll
            for (int i = 0; i < GBN * (GBK / 8) / 128; ++i) {
                int c = tid + i * 128;
                int r = c >> 2, kk = (c & 3) * 8;
                int gn = tileN + r, gk = k0 + kk;
                v8h val = {};
                if (gn < N && gk < K)
                    val = *(const v8h*)(Bt + (long)gn * ldb + gk);
                *(v8h*)(Bs + r * LDS_S + kk) = val;
            }
        }
        // prefetch next K tile into GL2 (global_prefetch_b8, speculative)
        if (k0 + GBK < K) {
            __builtin_prefetch(A  + (long)(tileM + (tid & 127)) * lda + k0 + GBK, 0, 1);
            __builtin_prefetch(Bt + (long)(tileN + (tid & 63)) * ldb + k0 + GBK, 0, 1);
        }
        __syncthreads();

        // ---- fragments per ISA 16-bit WMMA layouts ----
        F16x16 afr[4], bfr[2];
        #pragma unroll
        for (int mi = 0; mi < 4; ++mi) {
            const half_t* base = As + (waveM + mi * 16 + lrow) * LDS_S;
            afr[mi].h[0] = *(const v8h*)(base + hsel * 8);        // K 0..7 | 8..15
            afr[mi].h[1] = *(const v8h*)(base + 16 + hsel * 8);   // K 16..23 | 24..31
        }
        #pragma unroll
        for (int ni = 0; ni < 2; ++ni) {
            const half_t* base = Bs + (waveN + ni * 16 + lrow) * LDS_S;
            bfr[ni].h[0] = *(const v8h*)(base + hsel * 8);
            bfr[ni].h[1] = *(const v8h*)(base + 16 + hsel * 8);
        }
        #pragma unroll
        for (int mi = 0; mi < 4; ++mi)
            #pragma unroll
            for (int ni = 0; ni < 2; ++ni)
                acc[mi][ni] = __builtin_amdgcn_wmma_f32_16x16x32_f16(
                    false, afr[mi].v, false, bfr[ni].v,
                    (short)0, acc[mi][ni], false, false);

        __syncthreads();
    }

    // ---- store C: lanes 0-15 -> rows 0..7, lanes 16-31 -> rows 8..15 ----
    const int rbase = hsel * 8;
    #pragma unroll
    for (int mi = 0; mi < 4; ++mi) {
        #pragma unroll
        for (int ni = 0; ni < 2; ++ni) {
            int col = tileN + waveN + ni * 16 + lrow;
            if (!FULL && col >= N) continue;
            #pragma unroll
            for (int r = 0; r < 8; ++r) {
                int row = tileM + waveM + mi * 16 + rbase + r;
                if (FULL || row < M) {
                    long idx = (long)row * ldc + col;
                    float v = acc[mi][ni][r];
                    C[idx] = beta ? (C[idx] + v) : v;
                }
            }
        }
    }
}

// ---------------- elementwise / prep kernels -------------------------------

__global__ void cvt_f32_f16_kernel(const float* __restrict__ s, half_t* __restrict__ d, long n) {
    long i = (long)blockIdx.x * blockDim.x + threadIdx.x;
    if (i < n) d[i] = (half_t)s[i];
}

// src [R][C] f32 -> dst [C][R] f16
__global__ void transpose_cvt_kernel(const float* __restrict__ src, half_t* __restrict__ dst,
                                     int R, int C) {
    long i = (long)blockIdx.x * blockDim.x + threadIdx.x;
    if (i >= (long)R * C) return;
    int c = (int)(i % C), r = (int)(i / C);
    dst[(long)c * R + r] = (half_t)src[i];
}

// Q: l2norm per (b,n,h) over DH, * (1/max(exp(scale_p[h]),0.01)), -> [b,h,n,d] f16
__global__ void q_prep_kernel(const float* __restrict__ qf, const float* __restrict__ scale_p,
                              half_t* __restrict__ qh) {
    int idx = blockIdx.x * blockDim.x + threadIdx.x;   // B*N*H
    if (idx >= B_ * N_ * H_) return;
    int h = idx % H_, n = (idx / H_) % N_, b = idx / (H_ * N_);
    const float* src = qf + (long)(b * N_ + n) * (H_ * DH_) + h * DH_;
    float ss = 0.f;
    for (int d = 0; d < DH_; ++d) { float v = src[d]; ss += v * v; }
    float inv = 1.0f / fmaxf(sqrtf(ss), 1e-12f);
    float sc  = 1.0f / fmaxf(__expf(scale_p[h]), 0.01f);
    half_t* dst = qh + ((long)(b * H_ + h) * N_ + n) * DH_;
    for (int d = 0; d < DH_; ++d) dst[d] = (half_t)(src[d] * inv * sc);
}

// K: prepend mem_k, l2norm, -> [b,h,j,d] f16
__global__ void k_prep_kernel(const float* __restrict__ kf, const float* __restrict__ mem_k,
                              half_t* __restrict__ kh) {
    int idx = blockIdx.x * blockDim.x + threadIdx.x;   // B*H*J
    if (idx >= B_ * H_ * J_) return;
    int j = idx % J_, h = (idx / J_) % H_, b = idx / (J_ * H_);
    const float* src = (j < NMEM_)
        ? mem_k + ((long)h * NMEM_ + j) * DH_
        : kf + (long)(b * NCTX_ + (j - NMEM_)) * (H_ * DH_) + h * DH_;
    float ss = 0.f;
    for (int d = 0; d < DH_; ++d) { float v = src[d]; ss += v * v; }
    float inv = 1.0f / fmaxf(sqrtf(ss), 1e-12f);
    half_t* dst = kh + ((long)(b * H_ + h) * J_ + j) * DH_;
    for (int d = 0; d < DH_; ++d) dst[d] = (half_t)(src[d] * inv);
}

// V: prepend mem_v, store TRANSPOSED [b,h,d,j] f16 (NT form for attn@V)
__global__ void v_prep_kernel(const float* __restrict__ vf, const float* __restrict__ mem_v,
                              half_t* __restrict__ vt) {
    int idx = blockIdx.x * blockDim.x + threadIdx.x;   // B*H*J
    if (idx >= B_ * H_ * J_) return;
    int j = idx % J_, h = (idx / J_) % H_, b = idx / (J_ * H_);
    const float* src = (j < NMEM_)
        ? mem_v + ((long)h * NMEM_ + j) * DH_
        : vf + (long)(b * NCTX_ + (j - NMEM_)) * (H_ * DH_) + h * DH_;
    half_t* dst = vt + (long)(b * H_ + h) * DH_ * J_ + j;
    for (int d = 0; d < DH_; ++d) dst[(long)d * J_] = (half_t)src[d];
}

// talking-heads pre-mix, in place: one thread owns all 16 heads of one (b,i,j)
__global__ void headmix_pre_kernel(float* __restrict__ dots, const float* __restrict__ th) {
    const long npb = (long)N_ * J_;
    long idx = (long)blockIdx.x * blockDim.x + threadIdx.x;   // B * N * J
    if (idx >= (long)B_ * npb) return;
    int b = (int)(idx / npb);
    long ij = idx % npb;
    long base = (long)b * H_ * npb + ij;
    float vals[H_];
    #pragma unroll
    for (int h = 0; h < H_; ++h) vals[h] = dots[base + (long)h * npb];
    #pragma unroll
    for (int g = 0; g < H_; ++g) {
        float a = 0.f;
        #pragma unroll
        for (int h = 0; h < H_; ++h) a += th[g * H_ + h] * vals[h];
        dots[base + (long)g * npb] = a;
    }
}

// softmax over j, one wave32 per row
__global__ __launch_bounds__(256)
void softmax_rows_kernel(float* __restrict__ d) {
    int row  = blockIdx.x * 8 + (threadIdx.x >> 5);   // rows = B*H*N
    int lane = threadIdx.x & 31;
    float* p = d + (long)row * J_;
    float m = -1e30f;
    for (int j = lane; j < J_; j += 32) m = fmaxf(m, p[j]);
    for (int o = 16; o > 0; o >>= 1) m = fmaxf(m, __shfl_xor(m, o, 32));
    float s = 0.f;
    for (int j = lane; j < J_; j += 32) s += __expf(p[j] - m);
    for (int o = 16; o > 0; o >>= 1) s += __shfl_xor(s, o, 32);
    float inv = 1.0f / s;
    for (int j = lane; j < J_; j += 32) p[j] = __expf(p[j] - m) * inv;
}

// talking-heads post-mix -> f16 attn buffer
__global__ void headmix_post_kernel(const float* __restrict__ dots, const float* __restrict__ th,
                                    half_t* __restrict__ attn) {
    const long npb = (long)N_ * J_;
    long idx = (long)blockIdx.x * blockDim.x + threadIdx.x;
    if (idx >= (long)B_ * npb) return;
    int b = (int)(idx / npb);
    long ij = idx % npb;
    long base = (long)b * H_ * npb + ij;
    float vals[H_];
    #pragma unroll
    for (int h = 0; h < H_; ++h) vals[h] = dots[base + (long)h * npb];
    #pragma unroll
    for (int g = 0; g < H_; ++g) {
        float a = 0.f;
        #pragma unroll
        for (int h = 0; h < H_; ++h) a += th[g * H_ + h] * vals[h];
        attn[base + (long)g * npb] = (half_t)a;
    }
}

// o[b,h,i,d] f32 * hs[h] -> ao[b,i,h*d] f16
__global__ void assemble_kernel(const float* __restrict__ o, const float* __restrict__ hs,
                                half_t* __restrict__ ao) {
    long idx = (long)blockIdx.x * blockDim.x + threadIdx.x;   // B*H*N*DH
    if (idx >= (long)B_ * H_ * N_ * DH_) return;
    int d = (int)(idx % DH_);
    int n = (int)((idx / DH_) % N_);
    int h = (int)((idx / ((long)DH_ * N_)) % H_);
    int b = (int)(idx / ((long)DH_ * N_ * H_));
    float v = o[idx] * hs[h];
    ao[(long)(b * N_ + n) * (H_ * DH_) + h * DH_ + d] = (half_t)v;
}

// ---------------------------- host side ------------------------------------

static inline void launch_gemm(const half_t* A, const half_t* Bt, float* C,
                               int M, int N, int K,
                               long sA, long sB, long sC,
                               int lda, int ldb, int ldc,
                               int batch, int beta, hipStream_t stream) {
    dim3 grid((N + GBN - 1) / GBN, (M + GBM - 1) / GBM, batch);
    bool full = (M % GBM == 0) && (N % GBN == 0) && (K % GBK == 0);
    if (full)
        gemm_nt_kernel<true><<<grid, 128, 0, stream>>>(A, Bt, C, M, N, K, sA, sB, sC, lda, ldb, ldc, beta);
    else
        gemm_nt_kernel<false><<<grid, 128, 0, stream>>>(A, Bt, C, M, N, K, sA, sB, sC, lda, ldb, ldc, beta);
}

extern "C" void kernel_launch(void* const* d_in, const int* in_sizes, int n_in,
                              void* d_out, int out_size, void* d_ws, size_t ws_size,
                              hipStream_t stream) {
    (void)in_sizes; (void)n_in; (void)out_size; (void)ws_size;

    const float* x   = (const float*)d_in[0];
    const float* ctx = (const float*)d_in[1];
    float* out = (float*)d_out;

    char* ws = (char*)d_ws;
    size_t off = 0;
    auto alloc = [&](size_t bytes) -> void* {
        void* p = ws + off;
        off = (off + bytes + 255) & ~(size_t)255;
        return p;
    };

    const long MQ = (long)B_ * N_;          // 2048 query rows
    const long MC = (long)B_ * NCTX_;       // 4096 context rows
    const int  HD = H_ * DH_;               // 1024

    half_t* xh    = (half_t*)alloc(MQ * DIM_ * sizeof(half_t));
    half_t* ctxh  = (half_t*)alloc(MC * DIM_ * sizeof(half_t));
    half_t* wT    = (half_t*)alloc((size_t)DIM_ * HD * sizeof(half_t));   // reused (stream-ordered)
    float*  projQ = (float*)alloc(MQ * HD * sizeof(float));
    float*  projK = (float*)alloc(MC * HD * sizeof(float));
    float*  projV = (float*)alloc(MC * HD * sizeof(float));
    half_t* qh    = (half_t*)alloc((size_t)B_ * H_ * N_ * DH_ * sizeof(half_t));
    half_t* kh    = (half_t*)alloc((size_t)B_ * H_ * J_ * DH_ * sizeof(half_t));
    half_t* vt    = (half_t*)alloc((size_t)B_ * H_ * DH_ * J_ * sizeof(half_t));
    float*  dots  = (float*)alloc((size_t)B_ * H_ * N_ * J_ * sizeof(float));    // 270 MB
    half_t* attnh = (half_t*)alloc((size_t)B_ * H_ * N_ * J_ * sizeof(half_t));  // 135 MB
    float*  obuf  = (float*)alloc((size_t)B_ * H_ * N_ * DH_ * sizeof(float));
    half_t* ao    = (half_t*)alloc(MQ * HD * sizeof(half_t));

    // x -> f16 once
    {
        long n = MQ * DIM_;
        cvt_f32_f16_kernel<<<(int)((n + 255) / 256), 256, 0, stream>>>(x, xh, n);
    }

    for (int br = 0; br < 2; ++br) {
        const float* wq      = (const float*)d_in[2 + 3 * br];
        const float* wk      = (const float*)d_in[3 + 3 * br];
        const float* wv      = (const float*)d_in[4 + 3 * br];
        const float* th_pre  = (const float*)d_in[8 + 2 * br];
        const float* th_post = (const float*)d_in[9 + 2 * br];
        const float* scale_p = (const float*)d_in[12 + br];
        const float* mem_k   = (const float*)d_in[14 + 2 * br];
        const float* mem_v   = (const float*)d_in[15 + 2 * br];
        const float* hs      = (const float*)d_in[18 + br];
        const float* wo      = (const float*)d_in[20 + br];
        const float* ctx_b   = ctx + (long)br * MC * DIM_;

        // context[br] -> f16
        {
            long n = MC * DIM_;
            cvt_f32_f16_kernel<<<(int)((n + 255) / 256), 256, 0, stream>>>(ctx_b, ctxh, n);
        }

        const long nW = (long)DIM_ * HD;
        const int  gW = (int)((nW + 255) / 256);

        // Q projection
        transpose_cvt_kernel<<<gW, 256, 0, stream>>>(wq, wT, DIM_, HD);
        launch_gemm(xh, wT, projQ, (int)MQ, HD, DIM_, 0, 0, 0, DIM_, DIM_, HD, 1, 0, stream);
        // K projection
        transpose_cvt_kernel<<<gW, 256, 0, stream>>>(wk, wT, DIM_, HD);
        launch_gemm(ctxh, wT, projK, (int)MC, HD, DIM_, 0, 0, 0, DIM_, DIM_, HD, 1, 0, stream);
        // V projection
        transpose_cvt_kernel<<<gW, 256, 0, stream>>>(wv, wT, DIM_, HD);
        launch_gemm(ctxh, wT, projV, (int)MC, HD, DIM_, 0, 0, 0, DIM_, DIM_, HD, 1, 0, stream);

        // prep (norms, temperature, memory prepend, V transpose)
        q_prep_kernel<<<(B_ * N_ * H_ + 255) / 256, 256, 0, stream>>>(projQ, scale_p, qh);
        k_prep_kernel<<<(B_ * H_ * J_ + 255) / 256, 256, 0, stream>>>(projK, mem_k, kh);
        v_prep_kernel<<<(B_ * H_ * J_ + 255) / 256, 256, 0, stream>>>(projV, mem_v, vt);

        // dots = qh @ kh^T, batched over (b,h)   [guarded: N=2064]
        launch_gemm(qh, kh, dots, N_, J_, DH_,
                    (long)N_ * DH_, (long)J_ * DH_, (long)N_ * J_,
                    DH_, DH_, J_, B_ * H_, 0, stream);

        // talking heads + softmax
        {
            long n = (long)B_ * N_ * J_;
            int g = (int)((n + 255) / 256);
            headmix_pre_kernel<<<g, 256, 0, stream>>>(dots, th_pre);
            softmax_rows_kernel<<<(B_ * H_ * N_) / 8, 256, 0, stream>>>(dots);
            headmix_post_kernel<<<g, 256, 0, stream>>>(dots, th_post, attnh);
        }

        // o = attn @ v  (NT with vt = V^T), batched over (b,h)  [guarded: K=2064]
        launch_gemm(attnh, vt, obuf, N_, DH_, J_,
                    (long)N_ * J_, (long)DH_ * J_, (long)N_ * DH_,
                    J_, J_, DH_, B_ * H_, 0, stream);

        // head_scale + reorder to [b, n, h*d] f16
        assemble_kernel<<<(int)(((long)B_ * H_ * N_ * DH_ + 255) / 256), 256, 0, stream>>>(obuf, hs, ao);

        // out (+)= ao @ wo
        transpose_cvt_kernel<<<gW, 256, 0, stream>>>(wo, wT, HD, DIM_);
        launch_gemm(ao, wT, out, (int)MQ, DIM_, HD, 0, 0, 0, HD, HD, DIM_, 1, /*beta=*/br, stream);
    }
}
